// FourStreamGCN_Model_85529978733328
// MI455X (gfx1250) — compile-verified, matrix operand order checked by reading, not run on hardware
//
#include <hip/hip_runtime.h>
#include <hip/hip_bf16.h>
#include <math.h>

typedef __attribute__((ext_vector_type(16))) _Float16 v16h;
typedef __attribute__((ext_vector_type(8)))  _Float16 v8h;
typedef __attribute__((ext_vector_type(8)))  float    v8f;

#define GB  64
#define GT  150
#define GJ  25
#define GH  128
#define GBT (GB*GT)        /* 9600  */
#define GNB (GBT*GJ)       /* 240000 */
#define CW  3584           /* 25*128 + 3*128 */

/* ================= WMMA GEMM: C = act(A[M,K] @ W[N,K]^T + bias) (+res) ================
   - 256 threads = 8 waves, block tile 128x128, K-step 32 (one 16x16x32 WMMA per subtile)
   - f32 global -> f16 LDS conversion during staging, f32 accumulate
   - double-buffered LDS (ping-pong): next tile's global loads issue before WMMAs,
     conversion+LDS store after, one barrier per K-iteration
   - fast staging path: 4x global_load_b128 per operand per thread, two ds_store_b128 */
#define BM 128
#define BN 128
#define BK 32
#define LDT 40   /* padded LDS row stride in halves: 80B -> conflict-free 16B frag loads */

__global__ void __launch_bounds__(256)
wmma_gemm_kernel(const float* __restrict__ A, const float* __restrict__ W,
                 const float* __restrict__ bias, const float* __restrict__ res,
                 float* __restrict__ C, int M, int N, int K, int act)
{
  __shared__ alignas(16) _Float16 As[2][BM * LDT];
  __shared__ alignas(16) _Float16 Bs[2][BN * LDT];

  const int tid   = threadIdx.x;
  const int bm    = blockIdx.y * BM;
  const int bn    = blockIdx.x * BN;
  const int lane  = tid & 31;
  const int wave  = tid >> 5;
  const int wm    = wave >> 1;      /* 0..3 */
  const int wn    = wave & 1;       /* 0..1 */
  const int laneM = lane & 15;
  const int khalf = lane >> 4;

  v8f acc[2][4];
  #pragma unroll
  for (int i = 0; i < 2; i++)
    #pragma unroll
    for (int j = 0; j < 4; j++)
      #pragma unroll
      for (int e = 0; e < 8; e++) acc[i][j][e] = 0.f;

  /* staging geometry: 2 threads per row, 16 consecutive K-elements per thread */
  const int  ldrow = tid >> 1;
  const int  ldcb  = (tid & 1) * 16;
  const int  grA   = bm + ldrow;
  const int  grB   = bn + ldrow;
  const bool okA   = grA < M;
  const bool okB   = grB < N;
  const bool kvec  = ((K & 3) == 0);          /* row base 16B-aligned when K%4==0 */
  const float* ap  = A + (size_t)grA * K;
  const float* wp  = W + (size_t)grB * K;

  float ra[16], rb[16];

  auto load_tile = [&](int k0) {
    const int kb = k0 + ldcb;
    if (kvec && (kb + 16 <= K)) {
      if (okA) {
        const float4* p = (const float4*)(ap + kb);
        float4 t0 = p[0], t1 = p[1], t2 = p[2], t3 = p[3];
        ra[0]=t0.x; ra[1]=t0.y; ra[2]=t0.z; ra[3]=t0.w;
        ra[4]=t1.x; ra[5]=t1.y; ra[6]=t1.z; ra[7]=t1.w;
        ra[8]=t2.x; ra[9]=t2.y; ra[10]=t2.z; ra[11]=t2.w;
        ra[12]=t3.x; ra[13]=t3.y; ra[14]=t3.z; ra[15]=t3.w;
      } else {
        #pragma unroll
        for (int i = 0; i < 16; i++) ra[i] = 0.f;
      }
      if (okB) {
        const float4* p = (const float4*)(wp + kb);
        float4 t0 = p[0], t1 = p[1], t2 = p[2], t3 = p[3];
        rb[0]=t0.x; rb[1]=t0.y; rb[2]=t0.z; rb[3]=t0.w;
        rb[4]=t1.x; rb[5]=t1.y; rb[6]=t1.z; rb[7]=t1.w;
        rb[8]=t2.x; rb[9]=t2.y; rb[10]=t2.z; rb[11]=t2.w;
        rb[12]=t3.x; rb[13]=t3.y; rb[14]=t3.z; rb[15]=t3.w;
      } else {
        #pragma unroll
        for (int i = 0; i < 16; i++) rb[i] = 0.f;
      }
    } else {
      /* edge path: gather to registers first, convert later (loads overlap) */
      #pragma unroll
      for (int i = 0; i < 16; i++) {
        int k = kb + i;
        ra[i] = (okA && k < K) ? ap[k] : 0.f;
      }
      #pragma unroll
      for (int i = 0; i < 16; i++) {
        int k = kb + i;
        rb[i] = (okB && k < K) ? wp[k] : 0.f;
      }
    }
  };

  auto store_tile = [&](int buf) {
    v8h a0, a1, b0, b1;
    #pragma unroll
    for (int i = 0; i < 8; i++) {
      a0[i] = (_Float16)ra[i];  a1[i] = (_Float16)ra[8 + i];
      b0[i] = (_Float16)rb[i];  b1[i] = (_Float16)rb[8 + i];
    }
    *(v8h*)(&As[buf][ldrow * LDT + ldcb])     = a0;
    *(v8h*)(&As[buf][ldrow * LDT + ldcb + 8]) = a1;
    *(v8h*)(&Bs[buf][ldrow * LDT + ldcb])     = b0;
    *(v8h*)(&Bs[buf][ldrow * LDT + ldcb + 8]) = b1;
  };

  const int ktiles = (K + BK - 1) / BK;
  load_tile(0);
  store_tile(0);

  for (int kt = 0; kt < ktiles; kt++) {
    __syncthreads();
    if (kt + 1 < ktiles) load_tile((kt + 1) * BK);   /* issue next tile's loads */

    const int buf = kt & 1;
    v16h af[2], bf[4];
    #pragma unroll
    for (int sm = 0; sm < 2; sm++) {
      int r = wm * 32 + sm * 16 + laneM;
      v8h lo = *(const v8h*)(&As[buf][r * LDT + khalf * 8]);
      v8h hi = *(const v8h*)(&As[buf][r * LDT + 16 + khalf * 8]);
      af[sm] = __builtin_shufflevector(lo, hi, 0,1,2,3,4,5,6,7,8,9,10,11,12,13,14,15);
    }
    #pragma unroll
    for (int sn = 0; sn < 4; sn++) {
      int r = wn * 64 + sn * 16 + laneM;
      v8h lo = *(const v8h*)(&Bs[buf][r * LDT + khalf * 8]);
      v8h hi = *(const v8h*)(&Bs[buf][r * LDT + 16 + khalf * 8]);
      bf[sn] = __builtin_shufflevector(lo, hi, 0,1,2,3,4,5,6,7,8,9,10,11,12,13,14,15);
    }
    #pragma unroll
    for (int sm = 0; sm < 2; sm++)
      #pragma unroll
      for (int sn = 0; sn < 4; sn++)
        acc[sm][sn] = __builtin_amdgcn_wmma_f32_16x16x32_f16(
            false, af[sm], false, bf[sn], (short)0, acc[sm][sn], false, false);

    if (kt + 1 < ktiles) store_tile((kt + 1) & 1);   /* fill other buffer */
  }

  #pragma unroll
  for (int sm = 0; sm < 2; sm++)
    #pragma unroll
    for (int sn = 0; sn < 4; sn++)
      #pragma unroll
      for (int r = 0; r < 8; r++) {
        int grow = bm + wm * 32 + sm * 16 + r + 8 * khalf;
        int gcol = bn + wn * 64 + sn * 16 + laneM;
        if (grow < M && gcol < N) {
          float v = acc[sm][sn][r];
          if (bias) v += bias[gcol];
          if (act == 1)      v = fmaxf(v, 0.f);
          else if (act == 2) v = v / (1.f + __expf(-v));   /* SiLU */
          if (res) v += res[(size_t)grow * N + gcol];
          C[(size_t)grow * N + gcol] = v;
        }
      }
}

/* ================= small utility kernels ================ */
__global__ void fill_kernel(float* p, float v, size_t n) {
  size_t i = (size_t)blockIdx.x * 256 + threadIdx.x;
  if (i < n) p[i] = v;
}
__global__ void deg_edge_kernel(float* deg, const int* dst, int E) {
  int e = blockIdx.x * 256 + threadIdx.x;
  if (e < E) atomicAdd(&deg[dst[e]], 1.f);
}
__global__ void gcn_norm_kernel(const float* __restrict__ xw, const float* __restrict__ deg,
                                float* __restrict__ out, size_t n) {
  size_t idx = (size_t)blockIdx.x * 256 + threadIdx.x;
  if (idx < n * GH) out[idx] = xw[idx] / deg[idx >> 7];
}
__global__ void gcn_scatter_kernel(const float* __restrict__ xw, const float* __restrict__ deg,
                                   float* __restrict__ out, const int* __restrict__ src,
                                   const int* __restrict__ dst, int E) {
  int idx = blockIdx.x * 256 + threadIdx.x;
  if (idx < E * GH) {
    int e = idx >> 7, c = idx & 127;
    int s = src[e], d = dst[e];
    float nrm = rsqrtf(deg[s]) * rsqrtf(deg[d]);
    atomicAdd(&out[(size_t)d * GH + c], xw[(size_t)s * GH + c] * nrm);
  }
}
__global__ void bias_relu_kernel(float* x, const float* __restrict__ b, size_t n) {
  size_t idx = (size_t)blockIdx.x * 256 + threadIdx.x;
  if (idx < n * GH) x[idx] = fmaxf(x[idx] + b[idx & 127], 0.f);
}
__global__ void gen_sp_edges(int* src, int* dst) {
  int e = blockIdx.x * 256 + threadIdx.x;
  if (e < GJ * GJ) { src[e] = e / GJ; dst[e] = e % GJ; }
}

/* ================= hypergraph conv (per b*t sample, all in LDS) ================ */
__global__ void __launch_bounds__(256)
hyper_kernel(const float* __restrict__ sx, float* __restrict__ xv)
{
  __shared__ float xt[GJ * GH];
  __shared__ float xe[GJ * GH];
  __shared__ float Am[GJ * GJ];
  __shared__ float sq[GJ];
  __shared__ float cnt[GJ];
  __shared__ float red[256];
  __shared__ float thr;
  const int m = blockIdx.x, tid = threadIdx.x;
  const float* xm = sx + (size_t)m * GJ * GH;
  for (int i = tid; i < GJ * GH; i += 256) xt[i] = xm[i];
  __syncthreads();
  if (tid < GJ) {
    float s = 0.f;
    for (int c = 0; c < GH; c++) { float v = xt[tid * GH + c]; s += v * v; }
    sq[tid] = s;
  }
  __syncthreads();
  float lsum = 0.f;
  for (int p = tid; p < GJ * GJ; p += 256) {
    int n = p / GJ, v = p % GJ;
    float dot = 0.f;
    for (int c = 0; c < GH; c++) dot += xt[n * GH + c] * xt[v * GH + c];
    float d = sqrtf(fmaxf(sq[n] + sq[v] - 2.f * dot, 0.f));
    Am[p] = d; lsum += d;
  }
  red[tid] = lsum; __syncthreads();
  for (int s = 128; s > 0; s >>= 1) { if (tid < s) red[tid] += red[tid + s]; __syncthreads(); }
  if (tid == 0) thr = red[0] / (float)(GJ * GJ);
  __syncthreads();
  for (int p = tid; p < GJ * GJ; p += 256) Am[p] = (Am[p] < thr) ? 1.f : 0.f;
  __syncthreads();
  if (tid < GJ) {
    float s = 0.f;
    for (int v = 0; v < GJ; v++) s += Am[tid * GJ + v];
    cnt[tid] = s + 1e-6f;
  }
  __syncthreads();
  for (int idx = tid; idx < GJ * GH; idx += 256) {
    int e = idx >> 7, c = idx & 127;
    float s = 0.f;
    for (int v = 0; v < GJ; v++) s += Am[e * GJ + v] * xt[v * GH + c];
    xe[idx] = s / cnt[e];
  }
  __syncthreads();
  for (int idx = tid; idx < GJ * GH; idx += 256) {
    int n = idx >> 7, c = idx & 127;
    float s = 0.f;
    for (int e = 0; e < GJ; e++) s += Am[n * GJ + e] * xe[e * GH + c];
    xv[(size_t)m * GJ * GH + idx] = s / cnt[n];
  }
}

/* ================= MHA core (per (sample, head), scores+softmax in LDS) ================ */
__global__ void __launch_bounds__(128)
attn_kernel(const float* __restrict__ qkv, float* __restrict__ o)
{
  __shared__ float q[GJ * 32], k[GJ * 32], v[GJ * 32], ps[GJ * GJ];
  const int n = blockIdx.x >> 2, h = blockIdx.x & 3;
  const int tid = threadIdx.x;
  for (int idx = tid; idx < GJ * 32; idx += 128) {
    int s = idx >> 5, d = idx & 31;
    size_t base = ((size_t)(n * GJ + s)) * 384 + h * 32 + d;
    q[idx] = qkv[base]; k[idx] = qkv[base + 128]; v[idx] = qkv[base + 256];
  }
  __syncthreads();
  for (int pi = tid; pi < GJ * GJ; pi += 128) {
    int s = pi / GJ, t = pi % GJ;
    float dot = 0.f;
    for (int d = 0; d < 32; d++) dot += q[s * 32 + d] * k[t * 32 + d];
    ps[pi] = dot * 0.17677669529663687f;  /* 1/sqrt(32) */
  }
  __syncthreads();
  if (tid < GJ) {
    float mx = -1e30f;
    for (int t = 0; t < GJ; t++) mx = fmaxf(mx, ps[tid * GJ + t]);
    float sm = 0.f;
    for (int t = 0; t < GJ; t++) { float e = __expf(ps[tid * GJ + t] - mx); ps[tid * GJ + t] = e; sm += e; }
    float inv = 1.f / sm;
    for (int t = 0; t < GJ; t++) ps[tid * GJ + t] *= inv;
  }
  __syncthreads();
  for (int idx = tid; idx < GJ * 32; idx += 128) {
    int s = idx >> 5, d = idx & 31;
    float a = 0.f;
    for (int t = 0; t < GJ; t++) a += ps[s * GJ + t] * v[t * 32 + d];
    o[((size_t)(n * GJ + s)) * GH + h * 32 + d] = a;
  }
}

/* ================= layernorm over last dim C (power of 2, <=256) ================ */
__global__ void layernorm_kernel(const float* __restrict__ in, float* __restrict__ out,
                                 const float* __restrict__ g, const float* __restrict__ b,
                                 int C, int act)
{
  extern __shared__ float sred[];
  const int row = blockIdx.x, tid = threadIdx.x;
  float x = in[(size_t)row * C + tid];
  sred[tid] = x; __syncthreads();
  for (int s = C >> 1; s > 0; s >>= 1) { if (tid < s) sred[tid] += sred[tid + s]; __syncthreads(); }
  float mean = sred[0] / C; __syncthreads();
  sred[tid] = x * x; __syncthreads();
  for (int s = C >> 1; s > 0; s >>= 1) { if (tid < s) sred[tid] += sred[tid + s]; __syncthreads(); }
  float var = sred[0] / C - mean * mean;
  float y = (x - mean) * rsqrtf(var + 1e-5f) * g[tid] + b[tid];
  if (act == 1) y = fmaxf(y, 0.f);
  out[(size_t)row * C + tid] = y;
}

/* ================= comb assembly ================ */
__global__ void skel_gather_kernel(const float* __restrict__ ti, float* __restrict__ comb) {
  size_t idx = (size_t)blockIdx.x * 256 + threadIdx.x;
  if (idx >= (size_t)GBT * (GJ * GH)) return;
  size_t bt = idx / (GJ * GH); int jc = (int)(idx % (GJ * GH));
  int j = jc >> 7, c = jc & 127;
  comb[bt * CW + jc] = ti[((size_t)bt * GJ + j) * GH + c];
}
__global__ void jf_copy_kernel(const float* __restrict__ jf, float* __restrict__ comb) {
  int idx = blockIdx.x * 256 + threadIdx.x;
  if (idx >= GBT * GH) return;
  comb[(size_t)(idx >> 7) * CW + 3200 + (idx & 127)] = jf[idx];
}
__global__ void spmean_kernel(const float* __restrict__ sp, float* __restrict__ comb) {
  int idx = blockIdx.x * 256 + threadIdx.x;
  if (idx >= GBT * GH) return;
  int c = idx & 127, bt = idx >> 7;
  float s = 0.f;
  for (int j = 0; j < GJ; j++) s += sp[((size_t)bt * GJ + j) * GH + c];
  comb[(size_t)bt * CW + 3456 + c] = s * (1.f / GJ);
}

/* ================= temporal conv (pad=1, k=3) ================ */
__global__ void conv1_kernel(const float* __restrict__ x, const float* __restrict__ W,
                             const float* __restrict__ b, float* __restrict__ out) {
  int idx = blockIdx.x * 256 + threadIdx.x;
  if (idx >= GB * GH * GT) return;
  int t = idx % GT, o = (idx / GT) % GH, bb = idx / (GT * GH);
  float acc = b[o];
  for (int i = 0; i < 75; i++) {
    int j = i / 3, f = i % 3;
    const float* wr = W + (o * 75 + i) * 3;
    for (int kk = 0; kk < 3; kk++) {
      int tt = t + kk - 1;
      if (tt >= 0 && tt < GT)
        acc += wr[kk] * x[(((size_t)bb * GT + tt) * GJ + j) * 3 + f];
    }
  }
  out[idx] = fmaxf(acc, 0.f);
}
__global__ void conv2_kernel(const float* __restrict__ in, const float* __restrict__ W,
                             const float* __restrict__ b, float* __restrict__ comb) {
  int idx = blockIdx.x * 256 + threadIdx.x;
  if (idx >= GB * GT * GH) return;
  int o = idx % GH, t = (idx / GH) % GT, bb = idx / (GH * GT);
  float acc = b[o];
  for (int i = 0; i < GH; i++) {
    const float* wr = W + (o * GH + i) * 3;
    const float* ir = in + ((size_t)bb * GH + i) * GT;
    if (t > 0)      acc += wr[0] * ir[t - 1];
    acc += wr[1] * ir[t];
    if (t < GT - 1) acc += wr[2] * ir[t + 1];
  }
  comb[((size_t)bb * GT + t) * CW + 3328 + o] = fmaxf(acc, 0.f);
}

/* ================= GRU recurrence step (pre-gates gi computed by WMMA GEMM) ======== */
__global__ void __launch_bounds__(256)
gru_step_kernel(const float* __restrict__ gi, const float* __restrict__ Whh,
                const float* __restrict__ bhh, const float* __restrict__ hprev,
                float* __restrict__ hout, float* __restrict__ gseq, int t)
{
  __shared__ float hs[256];
  const int tid = threadIdx.x;
  const int g = blockIdx.x * 256 + tid;            /* 0 .. 8191 */
  hs[tid] = hprev[g];
  __syncthreads();
  const int b = g >> 7, i = g & 127;
  const float* hrow = &hs[(tid >> 7) << 7];
  const float* w0 = Whh + (size_t)i * GH;
  const float* w1 = Whh + (size_t)(GH + i) * GH;
  const float* w2 = Whh + (size_t)(2 * GH + i) * GH;
  float hr = 0.f, hz = 0.f, hg = 0.f;
  for (int k = 0; k < GH; k++) { float hv = hrow[k]; hr += w0[k] * hv; hz += w1[k] * hv; hg += w2[k] * hv; }
  hr += bhh[i]; hz += bhh[GH + i]; hg += bhh[2 * GH + i];
  size_t grow = (size_t)(b * GT + t) * 384;
  float ir = gi[grow + i], iz = gi[grow + GH + i], ig = gi[grow + 2 * GH + i];
  float r = 1.f / (1.f + __expf(-(ir + hr)));
  float z = 1.f / (1.f + __expf(-(iz + hz)));
  float nn = tanhf(ig + r * hg);
  float h = (1.f - z) * nn + z * hrow[i];
  hout[g] = h;
  gseq[(size_t)(b * GT + t) * GH + i] = h;
}

__global__ void fc_kernel(const float* __restrict__ h, const float* __restrict__ W,
                          const float* __restrict__ b, float* __restrict__ out) {
  int tid = threadIdx.x;
  if (tid < GB * 5) {
    int bb = tid / 5, o = tid % 5;
    float s = b[o];
    for (int k = 0; k < GH; k++) s += h[bb * GH + k] * W[o * GH + k];
    out[bb * 5 + o] = s;
  }
}

/* ================= host orchestration ================ */
static inline void launch_gemm(hipStream_t s, const float* A, const float* W, const float* bias,
                               const float* res, float* C, int M, int N, int K, int act) {
  dim3 g((N + BN - 1) / BN, (M + BM - 1) / BM);
  wmma_gemm_kernel<<<g, dim3(256), 0, s>>>(A, W, bias, res, C, M, N, K, act);
}

#define K1D(kern, n, ...) kern<<<dim3((unsigned)(((size_t)(n) + 255) / 256)), dim3(256), 0, stream>>>(__VA_ARGS__)

static void run_gcn(hipStream_t stream, const float* in, int Kdim, const float* Wt, const float* bt,
                    float* xw, float* out, float* deg, size_t n,
                    const int* srcp, const int* dstp, int E) {
  launch_gemm(stream, in, Wt, nullptr, nullptr, xw, (int)n, GH, Kdim, 0);
  K1D(fill_kernel, n, deg, 1.f, n);
  K1D(deg_edge_kernel, E, deg, dstp, E);
  K1D(gcn_norm_kernel, n * GH, xw, deg, out, n);
  K1D(gcn_scatter_kernel, (size_t)E * GH, xw, deg, out, srcp, dstp, E);
  K1D(bias_relu_kernel, n * GH, out, bt, n);
}

extern "C" void kernel_launch(void* const* d_in, const int* in_sizes, int n_in,
                              void* d_out, int out_size, void* d_ws, size_t ws_size,
                              hipStream_t stream)
{
  (void)in_sizes; (void)n_in; (void)out_size; (void)ws_size;
  const float* x   = (const float*)d_in[0];
  const int*   ei  = (const int*)d_in[1];
  const float* jcd = (const float*)d_in[2];
  const float* P[49];
  for (int i = 0; i < 49; i++) P[i] = (const float*)d_in[i];
  const float *sg1W=P[3], *sg1b=P[4], *sg2W=P[5], *sg2b=P[6];
  const float *hyW=P[7],  *hyb=P[8];
  const float *Wqkv=P[9], *bqkv=P[10], *Wo=P[11], *bo=P[12];
  const float *ln1g=P[13],*ln1b=P[14];
  const float *ff1W=P[15],*ff1b=P[16], *ff2W=P[17], *ff2b=P[18];
  const float *ln2g=P[19],*ln2b=P[20];
  const float *jg1W=P[21],*jg1b=P[22], *jg2W=P[23], *jg2b=P[24];
  const float *tc1W=P[25],*tc1b=P[26], *tc2W=P[27], *tc2b=P[28];
  const float *pg1W=P[29],*pg1b=P[30], *pg2W=P[31], *pg2b=P[32];
  const float *fu1W=P[33],*fu1b=P[34], *flng=P[35], *flnb=P[36];
  const float *fu2W=P[37],*fu2b=P[38];
  const float *g0Wih=P[39],*g0Whh=P[40],*g0bih=P[41],*g0bhh=P[42];
  const float *g1Wih=P[43],*g1Whh=P[44],*g1bih=P[45],*g1bhh=P[46];
  const float *fcW=P[47], *fcb=P[48];
  const int* e_src = ei;
  const int* e_dst = ei + 50;

  /* -------- workspace allocator -------- */
  char* wp = (char*)d_ws;
  auto alloc = [&](size_t bytes) -> float* {
    float* r = (float*)wp;
    wp += (bytes + 255) & ~(size_t)255;
    return r;
  };
  float* nodeA = alloc((size_t)GNB * GH * 4);
  float* nodeB = alloc((size_t)GNB * GH * 4);
  float* nodeC = alloc((size_t)GNB * GH * 4);
  float* bigQ  = alloc((size_t)GNB * 512 * 4);   /* qkv (384 cols) + FF chunk temp */
  float* degN  = alloc((size_t)GNB * 4);
  float* degJ  = alloc((size_t)GBT * 4);
  int*   spe   = (int*)alloc((size_t)2 * GJ * GJ * 4);
  float* comb  = alloc((size_t)GBT * CW * 4);
  float* c1    = alloc((size_t)GB * GH * GT * 4);
  float* fu1o  = alloc((size_t)GBT * 256 * 4);
  float* fz    = alloc((size_t)GBT * GH * 4);
  float* gib   = alloc((size_t)GBT * 384 * 4);
  float* gs0   = alloc((size_t)GBT * GH * 4);
  float* gs1   = alloc((size_t)GBT * GH * 4);
  float* hb0   = alloc((size_t)GB * GH * 4);
  float* hb1   = alloc((size_t)GB * GH * 4);
  float* jt1   = alloc((size_t)GBT * GH * 4);
  float* jt2   = alloc((size_t)GBT * GH * 4);

  /* -------- skeleton stream: two GCNs -------- */
  run_gcn(stream, x,     3,  sg1W, sg1b, nodeA, nodeB, degN, GNB, e_src, e_dst, 50);
  run_gcn(stream, nodeB, GH, sg2W, sg2b, nodeA, nodeC, degN, GNB, e_src, e_dst, 50);

  /* -------- hypergraph conv: xv -> GEMM(silu)+residual -------- */
  hyper_kernel<<<dim3(GBT), dim3(256), 0, stream>>>(nodeC, nodeA);
  launch_gemm(stream, nodeA, hyW, hyb, nodeC, nodeB, GNB, GH, GH, 2);   /* ti = silu(xv@W+b)+xt */

  /* -------- transformer block -------- */
  launch_gemm(stream, nodeB, Wqkv, bqkv, nullptr, bigQ, GNB, 384, GH, 0);
  attn_kernel<<<dim3(GBT * 4), dim3(128), 0, stream>>>(bigQ, nodeA);
  launch_gemm(stream, nodeA, Wo, bo, nodeB, nodeC, GNB, GH, GH, 0);     /* ti + attn */
  layernorm_kernel<<<dim3(GNB), dim3(GH), GH * 4, stream>>>(nodeC, nodeB, ln1g, ln1b, GH, 0);
  for (int ch = 0; ch < 2; ch++) {                                      /* FF chunked: reuse bigQ */
    size_t r0 = (size_t)ch * (GNB / 2);
    launch_gemm(stream, nodeB + r0 * GH, ff1W, ff1b, nullptr, bigQ, GNB / 2, 512, GH, 1);
    launch_gemm(stream, bigQ, ff2W, ff2b, nodeB + r0 * GH, nodeC + r0 * GH, GNB / 2, GH, 512, 0);
  }
  layernorm_kernel<<<dim3(GNB), dim3(GH), GH * 4, stream>>>(nodeC, nodeB, ln2g, ln2b, GH, 0);
  K1D(skel_gather_kernel, (size_t)GBT * GJ * GH, nodeB, comb);

  /* -------- spatial stream (dense JxJ edges) -------- */
  K1D(gen_sp_edges, GJ * GJ, spe, spe + GJ * GJ);
  run_gcn(stream, x,     3,  pg1W, pg1b, nodeA, nodeB, degN, GNB, spe, spe + GJ * GJ, GJ * GJ);
  run_gcn(stream, nodeB, GH, pg2W, pg2b, nodeA, nodeC, degN, GNB, spe, spe + GJ * GJ, GJ * GJ);
  K1D(spmean_kernel, GBT * GH, nodeC, comb);

  /* -------- JCD stream -------- */
  run_gcn(stream, jcd, 300, jg1W, jg1b, jt1, jt2, degJ, GBT, e_src, e_dst, 50);
  run_gcn(stream, jt2, GH,  jg2W, jg2b, jt1, jt2, degJ, GBT, e_src, e_dst, 50);
  K1D(jf_copy_kernel, GBT * GH, jt2, comb);

  /* -------- temporal conv stream -------- */
  K1D(conv1_kernel, GB * GH * GT, x, tc1W, tc1b, c1);
  K1D(conv2_kernel, GB * GT * GH, c1, tc2W, tc2b, comb);

  /* -------- fusion -------- */
  launch_gemm(stream, comb, fu1W, fu1b, nullptr, fu1o, GBT, 256, CW, 0);
  layernorm_kernel<<<dim3(GBT), dim3(256), 256 * 4, stream>>>(fu1o, fu1o, flng, flnb, 256, 1);
  launch_gemm(stream, fu1o, fu2W, fu2b, nullptr, fz, GBT, GH, 256, 0);

  /* -------- GRU layer 0 -------- */
  launch_gemm(stream, fz, g0Wih, g0bih, nullptr, gib, GBT, 384, GH, 0);
  K1D(fill_kernel, GB * GH, hb0, 0.f, (size_t)GB * GH);
  float *hc = hb0, *hn = hb1;
  for (int t = 0; t < GT; t++) {
    gru_step_kernel<<<dim3(32), dim3(256), 0, stream>>>(gib, g0Whh, g0bhh, hc, hn, gs0, t);
    float* tmp = hc; hc = hn; hn = tmp;
  }
  /* -------- GRU layer 1 -------- */
  launch_gemm(stream, gs0, g1Wih, g1bih, nullptr, gib, GBT, 384, GH, 0);
  hc = hb0; hn = hb1;
  K1D(fill_kernel, GB * GH, hb0, 0.f, (size_t)GB * GH);
  for (int t = 0; t < GT; t++) {
    gru_step_kernel<<<dim3(32), dim3(256), 0, stream>>>(gib, g1Whh, g1bhh, hc, hn, gs1, t);
    float* tmp = hc; hc = hn; hn = tmp;
  }
  /* after an even number of steps, final h lives in hb0 (== hc) */
  fc_kernel<<<dim3(1), dim3(320), 0, stream>>>(hc, fcW, fcb, (float*)d_out);
}
